// BiLSTM_CRF_85615878079152
// MI455X (gfx1250) — compile-verified, matrix operand order checked by reading, not run on hardware
//
#include <hip/hip_runtime.h>
#include <hip/hip_bf16.h>

typedef __attribute__((ext_vector_type(16))) __bf16 v16bf;
typedef __attribute__((ext_vector_type(8)))  __bf16 v8bf;
typedef __attribute__((ext_vector_type(8)))  float  v8f;

#define T_LEN  512
#define BATCH  64
#define EMB    256
#define HID    256
#define NSTATE 64
#define GATES  1024   // 4*HID
#define KDIM   512    // EMB + HID

__device__ __forceinline__ float sigf(float x) { return 1.0f / (1.0f + __expf(-x)); }

__device__ __forceinline__ v8f wmma_bf16(v16bf a, v16bf b, v8f c) {
  return __builtin_amdgcn_wmma_f32_16x16x32_bf16(false, a, false, b, (short)0, c, false, false);
}

__device__ __forceinline__ v16bf join16(v8bf p0, v8bf p1) {
  return __builtin_shufflevector(p0, p1, 0, 1, 2, 3, 4, 5, 6, 7,
                                 8, 9, 10, 11, 12, 13, 14, 15);
}

// ---------------- Kernel 1: embedding gather + f32 -> bf16 ----------------
__global__ void embed_kernel(const int* __restrict__ obs, const float* __restrict__ table,
                             __bf16* __restrict__ X) {
  int idx = blockIdx.x * blockDim.x + threadIdx.x;
  const int total = T_LEN * BATCH * EMB;
  for (; idx < total; idx += gridDim.x * blockDim.x) {
    int row = idx >> 8;          // t*B + b
    int col = idx & 255;
    int tok = obs[row];
    X[idx] = (__bf16)table[(size_t)tok * EMB + col];
  }
}

// ---------------- Kernel 2: pack weights into WMMA B-fragment order ----------------
// packedW layout: [dir][nTile(64)][kTile(16)][lane(32)][16 elems]  (bf16)
// Fragment element e of lane: k_local = (e<8 ? laneHi*8+e : 16 + laneHi*8 + (e-8))
__global__ void pack_kernel(const float* __restrict__ Wih_f, const float* __restrict__ Whh_f,
                            const float* __restrict__ bih_f, const float* __restrict__ bhh_f,
                            const float* __restrict__ Wih_b, const float* __restrict__ Whh_b,
                            const float* __restrict__ bih_b, const float* __restrict__ bhh_b,
                            const float* __restrict__ W_lin,
                            __bf16* __restrict__ packedW, __bf16* __restrict__ packedL,
                            float* __restrict__ biasg) {
  const int NW = 2 * 64 * 16 * 32;   // lane-sets for packedW
  const int NL = 4 * 16 * 32;        // lane-sets for packedL
  const int NB = 2 * GATES;
  int id = blockIdx.x * blockDim.x + threadIdx.x;
  if (id < NW) {
    int lane = id & 31, kt = (id >> 5) & 15, nT = (id >> 9) & 63, dir = id >> 15;
    int hi = lane >> 4, lo = lane & 15;
    int n = nT * 16 + lo;            // row of [Wih;Whh] (gate-major: i,f,g,o)
    const float* Wih = dir ? Wih_b : Wih_f;
    const float* Whh = dir ? Whh_b : Whh_f;
    __bf16* dst = packedW + (size_t)id * 16;
#pragma unroll
    for (int e = 0; e < 16; ++e) {
      int kl = (e < 8) ? (hi * 8 + e) : (16 + hi * 8 + (e - 8));
      int k = kt * 32 + kl;          // 0..511 (EMB then HID)
      float v = (k < EMB) ? Wih[(size_t)n * EMB + k] : Whh[(size_t)n * HID + (k - EMB)];
      dst[e] = (__bf16)v;
    }
  } else if (id < NW + NL) {
    int j = id - NW;
    int lane = j & 31, kt = (j >> 5) & 15, nT = j >> 9;
    int hi = lane >> 4, lo = lane & 15;
    int n = nT * 16 + lo;            // state row 0..63
    __bf16* dst = packedL + (size_t)j * 16;
#pragma unroll
    for (int e = 0; e < 16; ++e) {
      int kl = (e < 8) ? (hi * 8 + e) : (16 + hi * 8 + (e - 8));
      int k = kt * 32 + kl;          // 0..511
      dst[e] = (__bf16)W_lin[(size_t)n * KDIM + k];
    }
  } else if (id < NW + NL + NB) {
    int j = id - NW - NL;
    int dir = j >> 10, n = j & 1023;
    biasg[j] = dir ? (bih_b[n] + bhh_b[n]) : (bih_f[n] + bhh_f[n]);
  }
}

// ---------------- Kernel 3: fused recurrent BiLSTM (1 block per direction) ----------------
// 512 threads = 16 waves. Wave jG owns 16 hidden columns; computes i,f,g,o tiles for
// all 4 M-tiles (batch 64) with K = 512 = [x_t ; h_{t-1}].  f32 WMMA accumulation.
// h double-buffered in LDS -> single barrier per step; c lives in registers.
__global__ __launch_bounds__(512, 1)
void lstm_kernel(const __bf16* __restrict__ X, const __bf16* __restrict__ packedW,
                 const float* __restrict__ biasg, const float* __restrict__ h0,
                 const float* __restrict__ c0, __bf16* __restrict__ feats) {
  const int dir = blockIdx.x;
  __shared__ __align__(16) __bf16 h_lds[2][BATCH][HID];   // 64 KB double buffer
  const int tid = threadIdx.x;
  const int wave = tid >> 5, lane = tid & 31, lo = lane & 15, hi = lane >> 4;
  const int jG = wave;                                     // hidden cols jG*16..+15

  // init h state (f32 -> bf16) into buffer 0
  for (int i = tid; i < BATCH * HID; i += 512)
    h_lds[0][i >> 8][i & 255] = (__bf16)h0[(size_t)dir * BATCH * HID + i];

  // init c state: lane owns (b = mt*16 + hi*8 + r, col = jG*16 + lo) per D-fragment map
  float creg[4][8];
#pragma unroll
  for (int mt = 0; mt < 4; ++mt)
#pragma unroll
    for (int r = 0; r < 8; ++r)
      creg[mt][r] = c0[(size_t)dir * BATCH * HID + (size_t)(mt * 16 + hi * 8 + r) * HID + jG * 16 + lo];

  float bias[4];
#pragma unroll
  for (int g = 0; g < 4; ++g) bias[g] = biasg[dir * GATES + g * HID + jG * 16 + lo];

  const __bf16* wbase = packedW + (size_t)dir * 64 * 8192;  // 8192 bf16 per nTile
  __syncthreads();

  for (int s = 0; s < T_LEN; ++s) {
    const int t  = dir ? (T_LEN - 1 - s) : s;
    const int rb = s & 1, wb = (s + 1) & 1;

    // prefetch next step's 32KB X tile into cache while we compute this step
    if (s + 1 < T_LEN) {
      const int tn = dir ? (T_LEN - 2 - s) : (s + 1);
      __builtin_prefetch(X + (size_t)tn * BATCH * EMB + (size_t)tid * 32, 0, 3);
    }

    v8f acc[4][4];
#pragma unroll
    for (int g = 0; g < 4; ++g)
#pragma unroll
      for (int mt = 0; mt < 4; ++mt) acc[g][mt] = 0.0f;

    const __bf16* Xt = X + (size_t)t * BATCH * EMB;

    // ---- K = 0..255 : x_t from global (L2-resident) ----
#pragma unroll 1
    for (int kt = 0; kt < 8; ++kt) {
      v16bf a[4];
#pragma unroll
      for (int mt = 0; mt < 4; ++mt) {
        const __bf16* src = Xt + (size_t)(mt * 16 + lo) * EMB + kt * 32;
        v8bf p0 = *(const v8bf*)(src + hi * 8);
        v8bf p1 = *(const v8bf*)(src + 16 + hi * 8);
        a[mt] = join16(p0, p1);
      }
#pragma unroll
      for (int g = 0; g < 4; ++g) {
        const v16bf bf = *(const v16bf*)(wbase + (size_t)(g * 16 + jG) * 8192 +
                                         (size_t)(kt * 32 + lane) * 16);
#pragma unroll
        for (int mt = 0; mt < 4; ++mt) acc[g][mt] = wmma_bf16(a[mt], bf, acc[g][mt]);
      }
    }

    // ---- K = 256..511 : h_{t-1} from LDS (ds_load path) ----
#pragma unroll 1
    for (int kt = 0; kt < 8; ++kt) {
      v16bf a[4];
#pragma unroll
      for (int mt = 0; mt < 4; ++mt) {
        const __bf16* src = &h_lds[rb][mt * 16 + lo][kt * 32];
        v8bf p0 = *(const v8bf*)(src + hi * 8);
        v8bf p1 = *(const v8bf*)(src + 16 + hi * 8);
        a[mt] = join16(p0, p1);
      }
#pragma unroll
      for (int g = 0; g < 4; ++g) {
        const v16bf bf = *(const v16bf*)(wbase + (size_t)(g * 16 + jG) * 8192 +
                                         (size_t)((kt + 8) * 32 + lane) * 16);
#pragma unroll
        for (int mt = 0; mt < 4; ++mt) acc[g][mt] = wmma_bf16(a[mt], bf, acc[g][mt]);
      }
    }

    // ---- gates -> (h, c); write h to the OTHER LDS buffer, no barrier needed before ----
#pragma unroll
    for (int mt = 0; mt < 4; ++mt) {
#pragma unroll
      for (int r = 0; r < 8; ++r) {
        float iv = acc[0][mt][r] + bias[0];
        float fv = acc[1][mt][r] + bias[1];
        float gv = acc[2][mt][r] + bias[2];
        float ov = acc[3][mt][r] + bias[3];
        float cn = sigf(fv) * creg[mt][r] + sigf(iv) * tanhf(gv);
        creg[mt][r] = cn;
        float hv = sigf(ov) * tanhf(cn);
        int b   = mt * 16 + hi * 8 + r;
        int col = jG * 16 + lo;
        __bf16 hb = (__bf16)hv;
        h_lds[wb][b][col] = hb;
        feats[((size_t)t * BATCH + b) * KDIM + dir * HID + col] = hb;  // concat [fwd,bwd]
      }
    }
    __syncthreads();   // single barrier per step (double-buffered h)
  }
}

// ---------------- Kernel 4: emission scores GEMM [32768,512] x [512,64] ----------------
__global__ __launch_bounds__(256)
void scores_kernel(const __bf16* __restrict__ feats, const __bf16* __restrict__ packedL,
                   const float* __restrict__ b_lin, float* __restrict__ scores) {
  const int tid = threadIdx.x;
  const int wave = tid >> 5, lane = tid & 31, lo = lane & 15, hi = lane >> 4;
  const int mTile = blockIdx.x * 8 + wave;
  const size_t rowBase = (size_t)mTile * 16;
  v8f acc[4];
#pragma unroll
  for (int nT = 0; nT < 4; ++nT) acc[nT] = 0.0f;

#pragma unroll 1
  for (int kt = 0; kt < 16; ++kt) {
    const __bf16* src = feats + (rowBase + lo) * KDIM + kt * 32;
    v8bf p0 = *(const v8bf*)(src + hi * 8);
    v8bf p1 = *(const v8bf*)(src + 16 + hi * 8);
    v16bf a = join16(p0, p1);
#pragma unroll
    for (int nT = 0; nT < 4; ++nT) {
      const v16bf bf = *(const v16bf*)(packedL + (size_t)((nT * 16 + kt) * 32 + lane) * 16);
      acc[nT] = wmma_bf16(a, bf, acc[nT]);
    }
  }
#pragma unroll
  for (int nT = 0; nT < 4; ++nT)
#pragma unroll
    for (int r = 0; r < 8; ++r) {
      size_t grow = rowBase + r + hi * 8;
      int scol = nT * 16 + lo;
      scores[grow * NSTATE + scol] = acc[nT][r] + b_lin[scol];
    }
}

// ---------------- Kernel 5: Viterbi DP + backtrace (1 block per batch row) ----------------
__global__ __launch_bounds__(64)
void viterbi_kernel(const float* __restrict__ scores, const float* __restrict__ pairwise,
                    const float* __restrict__ startv, const float* __restrict__ stopv,
                    unsigned char* __restrict__ bps, float* __restrict__ out) {
  const int b = blockIdx.x, s = threadIdx.x;
  __shared__ float pw[NSTATE][NSTATE];   // 16 KB
  __shared__ float dbuf[2][NSTATE];
  __shared__ float fin[NSTATE];
  for (int i = s; i < NSTATE * NSTATE; i += NSTATE) pw[i >> 6][i & 63] = pairwise[i];
  dbuf[0][s] = startv[s] + scores[(size_t)b * NSTATE + s];
  __syncthreads();

  for (int t = 1; t < T_LEN; ++t) {
    const float* delta = dbuf[(t - 1) & 1];
    float best = -3.4e38f; int bp = 0;
    for (int p = 0; p < NSTATE; ++p) {           // ascending + strict '>' == jnp.argmax
      float v = delta[p] + pw[p][s];
      if (v > best) { best = v; bp = p; }
    }
    bps[((size_t)(t - 1) * BATCH + b) * NSTATE + s] = (unsigned char)bp;
    dbuf[t & 1][s] = best + scores[((size_t)t * BATCH + b) * NSTATE + s];
    __syncthreads();
  }

  fin[s] = dbuf[(T_LEN - 1) & 1][s] + stopv[s];
  __syncthreads();
  if (s == 0) {
    float best = fin[0]; int arg = 0;
    for (int i = 1; i < NSTATE; ++i) if (fin[i] > best) { best = fin[i]; arg = i; }
    out[b] = best;                                        // max_score
    int st = arg;
    out[NSTATE + (size_t)(T_LEN - 1) * BATCH + b] = (float)st;
    for (int t = T_LEN - 2; t >= 0; --t) {
      st = bps[((size_t)t * BATCH + b) * NSTATE + st];
      out[NSTATE + (size_t)t * BATCH + b] = (float)st;    // state_seq[t]
    }
  }
}

// ---------------- host launcher ----------------
extern "C" void kernel_launch(void* const* d_in, const int* in_sizes, int n_in,
                              void* d_out, int out_size, void* d_ws, size_t ws_size,
                              hipStream_t stream) {
  const int*   obs      = (const int*)d_in[0];
  const float* h0       = (const float*)d_in[1];
  const float* c0       = (const float*)d_in[2];
  const float* table    = (const float*)d_in[3];
  const float* Wih_f    = (const float*)d_in[4];
  const float* Whh_f    = (const float*)d_in[5];
  const float* bih_f    = (const float*)d_in[6];
  const float* bhh_f    = (const float*)d_in[7];
  const float* Wih_b    = (const float*)d_in[8];
  const float* Whh_b    = (const float*)d_in[9];
  const float* bih_b    = (const float*)d_in[10];
  const float* bhh_b    = (const float*)d_in[11];
  const float* W_lin    = (const float*)d_in[12];
  const float* b_lin    = (const float*)d_in[13];
  const float* pairwise = (const float*)d_in[14];
  const float* startv   = (const float*)d_in[15];
  const float* stopv    = (const float*)d_in[16];
  (void)in_sizes; (void)n_in; (void)out_size; (void)ws_size;

  char* ws = (char*)d_ws;
  size_t off = 0;
  auto alloc = [&](size_t bytes) -> char* {
    char* p = ws + off;
    off = (off + bytes + 255) & ~(size_t)255;
    return p;
  };
  __bf16* X       = (__bf16*)alloc(sizeof(__bf16) * (size_t)T_LEN * BATCH * EMB);      // 16 MB
  __bf16* packedW = (__bf16*)alloc(sizeof(__bf16) * (size_t)2 * 64 * 16 * 32 * 16);    // 2 MB
  __bf16* packedL = (__bf16*)alloc(sizeof(__bf16) * (size_t)4 * 16 * 32 * 16);         // 64 KB
  float*  biasg   = (float*) alloc(sizeof(float)  * 2 * GATES);
  __bf16* feats   = (__bf16*)alloc(sizeof(__bf16) * (size_t)T_LEN * BATCH * KDIM);     // 32 MB
  float*  scoresb = (float*) alloc(sizeof(float)  * (size_t)T_LEN * BATCH * NSTATE);   // 8 MB
  unsigned char* bps = (unsigned char*)alloc((size_t)(T_LEN - 1) * BATCH * NSTATE);    // 2 MB

  embed_kernel<<<1024, 256, 0, stream>>>(obs, table, X);

  const int packThreads = 2 * 64 * 16 * 32 + 4 * 16 * 32 + 2 * GATES;
  pack_kernel<<<(packThreads + 255) / 256, 256, 0, stream>>>(
      Wih_f, Whh_f, bih_f, bhh_f, Wih_b, Whh_b, bih_b, bhh_b, W_lin,
      packedW, packedL, biasg);

  lstm_kernel<<<2, 512, 0, stream>>>(X, packedW, biasg, h0, c0, feats);

  scores_kernel<<<256, 256, 0, stream>>>(feats, packedL, b_lin, scoresb);

  viterbi_kernel<<<BATCH, NSTATE, 0, stream>>>(scoresb, pairwise, startv, stopv, bps,
                                               (float*)d_out);
}
